// TransformerSwapGNN_44538810859551
// MI455X (gfx1250) — compile-verified, matrix operand order checked by reading, not run on hardware
//
#include <hip/hip_runtime.h>

#define NNODES 100000
#define NEDGES 800000

typedef unsigned short u16;
typedef unsigned int   u32;
typedef unsigned long long u64;
typedef __attribute__((ext_vector_type(16))) __bf16 v16bf;
typedef __attribute__((ext_vector_type(8)))  float  v8f;
typedef __attribute__((ext_vector_type(4)))  unsigned int v4u;
typedef __attribute__((ext_vector_type(8)))  int    v8i;
typedef __attribute__((ext_vector_type(4)))  int    v4i;

// ---------------- helpers ----------------
__device__ __forceinline__ u16 f2bf(float f){              // fp32 -> bf16, RNE
  u32 u = __float_as_uint(f);
  u32 r = u + 0x7FFFu + ((u >> 16) & 1u);
  return (u16)(r >> 16);
}
__device__ __forceinline__ u32 encf(float x){              // order-preserving f32->u32
  u32 u = __float_as_uint(x);
  return (u & 0x80000000u) ? ~u : (u | 0x80000000u);
}
__device__ __forceinline__ float decf(u32 u){
  u32 v = (u & 0x80000000u) ? (u & 0x7FFFFFFFu) : ~u;
  return __uint_as_float(v);
}

#define TF_RND(r) { x0 += x1; x1 = (x1<<(r))|(x1>>(32-(r))); x1 ^= x0; }
__host__ __device__ inline void threefry2x32(u32 k0,u32 k1,u32 c0,u32 c1,u32* o0,u32* o1){
  u32 ks2 = k0 ^ k1 ^ 0x1BD11BDAu;
  u32 x0 = c0 + k0, x1 = c1 + k1;
  TF_RND(13) TF_RND(15) TF_RND(26) TF_RND(6)  x0+=k1;  x1+=ks2+1u;
  TF_RND(17) TF_RND(29) TF_RND(16) TF_RND(24) x0+=ks2; x1+=k0+2u;
  TF_RND(13) TF_RND(15) TF_RND(26) TF_RND(6)  x0+=k0;  x1+=k1+3u;
  TF_RND(17) TF_RND(29) TF_RND(16) TF_RND(24) x0+=k1;  x1+=ks2+4u;
  TF_RND(13) TF_RND(15) TF_RND(26) TF_RND(6)  x0+=ks2; x1+=k0+5u;
  *o0 = x0; *o1 = x1;
}
__device__ __forceinline__ float bits_to_gumbel(u32 w){
  float u = __uint_as_float((w >> 9) | 0x3F800000u) - 1.0f;   // [0,1)
  u = fmaxf(u, 1.1754944e-38f);
  float l = fmaxf(-__logf(u), 1.1754944e-38f);
  return -__logf(l);
}

union V16u { uint4 q[2]; v16bf v; };
__device__ __forceinline__ v16bf frag_load(const u16* p, int kb){
  // CDNA5 16-bit A/B fragment: lane holds K in {kb..kb+7} and {kb+16..kb+23}
  V16u t;
  t.q[0] = *(const uint4*)(p + kb);
  t.q[1] = *(const uint4*)(p + kb + 16);
  return t.v;
}

// ---- Tensor Data Mover: 1-D copy of nelem bf16 words Global -> LDS ----
// D# per CDNA5 ISA: group0 = {count=1, lds_addr,
// global_addr, type=2}; group1 = {data_size=1(2B), dim0=tile0=nelem,
// dim1=tile1=1, stride0=nelem}; groups 2/3 zero (<=2-D mode).
__device__ __forceinline__ void tdm_load_1d(const void* gsrc, u16* lds_dst, u32 nelem){
  u64 ga = (u64)(size_t)gsrc;
  u32 la = (u32)(size_t)(void*)lds_dst;        // low 32 bits of LDS flat addr = byte offset
  v4u g0;
  g0[0] = 1u;                                  // count=1, user descriptor
  g0[1] = la;                                  // lds_addr (bytes)
  g0[2] = (u32)ga;                             // global_addr[31:0]
  g0[3] = (u32)((ga >> 32) & 0x01FFFFFFu) | (2u << 30);  // global_addr[56:32] | type=2
  v8i g1;
  g1[0] = (int)(1u << 16);                     // data_size=1 (2 bytes/element)
  g1[1] = (int)((nelem & 0xFFFFu) << 16);      // tensor_dim0[15:0]  @ bits 63:48
  g1[2] = (int)((nelem >> 16) | (1u << 16));   // tensor_dim0[31:16]; tensor_dim1=1
  g1[3] = (int)((nelem & 0xFFFFu) << 16);      // tile_dim0 @ bits 127:112
  g1[4] = 1;                                   // tile_dim1=1; tile_dim2=0
  g1[5] = (int)nelem;                          // tensor_dim0_stride[31:0]
  g1[6] = 0;
  g1[7] = 0;
  v4i z4 = {0,0,0,0};
#if defined(__clang_major__) && (__clang_major__ >= 23)
  v8i z8 = {0,0,0,0,0,0,0,0};
  __builtin_amdgcn_tensor_load_to_lds(g0, g1, z4, z4, z8, 0);   // clang-23 form
#else
  __builtin_amdgcn_tensor_load_to_lds(g0, g1, z4, z4, 0);       // ROCm 7.2 clang-22 form
#endif
}

// ---------------- utility kernels ----------------
__global__ void tsg_zero_f(float* p, size_t n){
  size_t i = (size_t)blockIdx.x*blockDim.x + threadIdx.x;
  if (i < n) p[i] = 0.f;
}
__global__ void tsg_zero_u(u32* p, size_t n){
  size_t i = (size_t)blockIdx.x*blockDim.x + threadIdx.x;
  if (i < n) p[i] = 0u;
}
__global__ void tsg_f2bf_k(const float* __restrict__ s, u16* __restrict__ d, int n){
  int i = blockIdx.x*blockDim.x + threadIdx.x;
  if (i < n) d[i] = f2bf(s[i]);
}

// ---------------- layer-1 direct projections (K=2, VALU is optimal) ----------------
__global__ void tsg_l1proj(const float* __restrict__ nt, const float* __restrict__ rq,
  const float* __restrict__ qw, const float* __restrict__ qb,
  const float* __restrict__ kw, const float* __restrict__ kb,
  const float* __restrict__ vw, const float* __restrict__ vb,
  const float* __restrict__ sw, const float* __restrict__ sb,
  float* __restrict__ q, float* __restrict__ k, float* __restrict__ v,
  float* __restrict__ skip){
  int i = blockIdx.x*blockDim.x + threadIdx.x;
  if (i >= NNODES*128) return;
  int n = i >> 7, o = i & 127;
  float x0 = nt[n], x1 = rq[n];
  q[i]    = fmaf(qw[2*o], x0, fmaf(qw[2*o+1], x1, qb[o]));
  k[i]    = fmaf(kw[2*o], x0, fmaf(kw[2*o+1], x1, kb[o]));
  v[i]    = fmaf(vw[2*o], x0, fmaf(vw[2*o+1], x1, vb[o]));
  skip[i] = fmaf(sw[2*o], x0, fmaf(sw[2*o+1], x1, sb[o]));
}

// ---------------- WMMA GEMM: out[M x Nc] = act(A[MxK]bf16 * W[NcxK]bf16^T + bias)
// Weight tile (K*Nc <= 8192 bf16) is DMA'd once per workgroup into LDS by the
// Tensor Data Mover; 8 waves then read B fragments via ds_load_b128. A tiles
// stream global->VGPR (per-wave unique).
__global__ void tsg_gemm(const u16* __restrict__ A, const u16* __restrict__ W,
                         const float* __restrict__ bias, float* __restrict__ out,
                         u16* __restrict__ outbf, int M, int K, int Nc, int relu){
  __shared__ __attribute__((aligned(16))) u16 wsm[8192];
  if (threadIdx.x < 32){                          // wave 0 issues the DMA
    tdm_load_1d(W, wsm, (u32)(K*Nc));
    __builtin_amdgcn_s_wait_tensorcnt(0);         // TENSORcnt is per-wave
  }
  __syncthreads();                                // publish LDS to all 8 waves
  int wave = (int)(((unsigned)blockIdx.x*blockDim.x + threadIdx.x) >> 5);
  int lane = threadIdx.x & 31;
  int ntiles = Nc >> 4;
  int mt = wave / ntiles;
  int nt = wave - mt*ntiles;
  if (mt*16 >= M) return;
  int half = lane >> 4, l15 = lane & 15, kb = half*8;
  const u16* arow = A + (size_t)(mt*16 + l15)*K;
  const u16* wrow = wsm + (size_t)(nt*16 + l15)*K;
  v8f c = {};
  for (int kt = 0; kt < K; kt += 32){
    v16bf a = frag_load(arow + kt, kb);
    v16bf b = frag_load(wrow + kt, kb);
    c = __builtin_amdgcn_wmma_f32_16x16x32_bf16(false, a, false, b, (short)0, c, false, false);
  }
  int nn = nt*16 + l15;
  float bv = bias[nn];
  #pragma unroll
  for (int r = 0; r < 8; r++){
    int mm = mt*16 + r + half*8;           // C layout: VGPR r -> M = r + 8*(lane>=16)
    float val = c[r] + bv;
    if (relu) val = fmaxf(val, 0.f);
    size_t idx = (size_t)mm*Nc + nn;
    out[idx] = val;
    if (outbf) outbf[idx] = f2bf(val);
  }
}

// ---------------- edge attention ----------------
__global__ void tsg_alpha(const int* __restrict__ src, const int* __restrict__ dst,
  const float* __restrict__ lat, const float* __restrict__ q, const float* __restrict__ k,
  const float* __restrict__ ew, const float* __restrict__ v,
  float* __restrict__ alpha, u32* __restrict__ amax){
  int i = blockIdx.x*blockDim.x + threadIdx.x;
  if (i >= NEDGES*4) return;
  int e = i >> 2, h = i & 3;
  int s = src[e], d = dst[e];
  float l = lat[e];
  __builtin_prefetch(v + (size_t)s*128 + h*32, 0, 1);   // pre-warm L2 for the message pass
  const float4* qp = (const float4*)(q + (size_t)d*128 + h*32);
  const float4* kp = (const float4*)(k + (size_t)s*128 + h*32);
  const float4* ep = (const float4*)(ew + h*32);
  float acc = 0.f;
  #pragma unroll
  for (int j = 0; j < 8; j++){
    float4 qv = qp[j], kv = kp[j], evv = ep[j];
    acc += qv.x*fmaf(l,evv.x,kv.x) + qv.y*fmaf(l,evv.y,kv.y)
         + qv.z*fmaf(l,evv.z,kv.z) + qv.w*fmaf(l,evv.w,kv.w);
  }
  float a = acc * 0.1767766952966369f;   // 1/sqrt(C)
  alpha[i] = a;
  atomicMax(&amax[(size_t)d*4 + h], encf(a));
}

__global__ void tsg_expnorm(const int* __restrict__ dst, const float* __restrict__ alpha,
  const u32* __restrict__ amax, float* __restrict__ denom, float* __restrict__ exb){
  int i = blockIdx.x*blockDim.x + threadIdx.x;
  if (i >= NEDGES*4) return;
  int e = i >> 2, h = i & 3;
  int d = dst[e];
  float ax = decf(amax[(size_t)d*4 + h]);
  float ev = __expf(alpha[i] - ax);
  exb[i] = ev;
  atomicAdd(&denom[(size_t)d*4 + h], ev);
}

__global__ void tsg_msg(const int* __restrict__ src, const int* __restrict__ dst,
  const float* __restrict__ lat, const float* __restrict__ v, const float* __restrict__ ew,
  const float* __restrict__ exb, const float* __restrict__ denom, float* __restrict__ agg){
  int t = blockIdx.x*blockDim.x + threadIdx.x;
  int e = t >> 5, lane = t & 31;                 // one wave32 per edge, float4 per lane
  if (e >= NEDGES) return;
  int s = src[e], d = dst[e];
  float l = lat[e];
  int h = lane >> 3;
  float dn = denom[(size_t)d*4 + h];
  float at = exb[(size_t)e*4 + h] / (dn > 0.f ? dn : 1.f);
  float4 vv = ((const float4*)(v + (size_t)s*128))[lane];
  float4 ee = ((const float4*)ew)[lane];
  float* ap = agg + (size_t)d*128 + lane*4;
  atomicAdd(ap+0, fmaf(l, ee.x, vv.x)*at);
  atomicAdd(ap+1, fmaf(l, ee.y, vv.y)*at);
  atomicAdd(ap+2, fmaf(l, ee.z, vv.z)*at);
  atomicAdd(ap+3, fmaf(l, ee.w, vv.w)*at);
}

// ---------------- beta gating: x = beta*skip + (1-beta)*out, emit bf16 ----------------
__global__ void tsg_gate(const float* __restrict__ agg, const float* __restrict__ skip,
  const float* __restrict__ bw, u16* __restrict__ xbf){
  int t = blockIdx.x*blockDim.x + threadIdx.x;
  int node = t >> 5, lane = t & 31;
  if (node >= NNODES) return;
  float4 o = ((const float4*)(agg  + (size_t)node*128))[lane];
  float4 s = ((const float4*)(skip + (size_t)node*128))[lane];
  const float4* b4 = (const float4*)bw;
  float4 b0 = b4[lane], b1 = b4[32+lane], b2 = b4[64+lane];
  float p = b0.x*o.x + b0.y*o.y + b0.z*o.z + b0.w*o.w
          + b1.x*s.x + b1.y*s.y + b1.z*s.z + b1.w*s.w
          + b2.x*(o.x-s.x) + b2.y*(o.y-s.y) + b2.z*(o.z-s.z) + b2.w*(o.w-s.w);
  #pragma unroll
  for (int off = 16; off; off >>= 1) p += __shfl_xor(p, off, 32);
  float beta = 1.f/(1.f + __expf(-p));
  union { u16 h4[4]; uint2 u2; } pk;
  pk.h4[0] = f2bf(beta*s.x + (1.f-beta)*o.x);
  pk.h4[1] = f2bf(beta*s.y + (1.f-beta)*o.y);
  pk.h4[2] = f2bf(beta*s.z + (1.f-beta)*o.z);
  pk.h4[3] = f2bf(beta*s.w + (1.f-beta)*o.w);
  *(uint2*)(xbf + (size_t)node*128 + lane*4) = pk.u2;
}

// ---------------- batch norm ----------------
__global__ void tsg_bnstats(const float* __restrict__ x, float* __restrict__ sums){
  __shared__ float ls[256], lq[256];
  int ch = threadIdx.x & 31, rg = threadIdx.x >> 5;
  float s = 0.f, qq = 0.f;
  for (int row = blockIdx.x*8 + rg; row < NNODES; row += gridDim.x*8){
    float vv = x[(size_t)row*32 + ch];
    s += vv; qq += vv*vv;
  }
  ls[threadIdx.x] = s; lq[threadIdx.x] = qq;
  __syncthreads();
  if (rg == 0){
    #pragma unroll
    for (int r = 1; r < 8; r++){ s += ls[r*32+ch]; qq += lq[r*32+ch]; }
    atomicAdd(&sums[ch], s);
    atomicAdd(&sums[32+ch], qq);
  }
}
__global__ void tsg_bnapply(float* __restrict__ x, const float* __restrict__ sums,
  const float* __restrict__ g, const float* __restrict__ b, u16* __restrict__ xbf){
  int i = blockIdx.x*blockDim.x + threadIdx.x;
  if (i >= NNODES*32) return;
  int ch = i & 31;
  const float invN = 1.0f/(float)NNODES;
  float mu  = sums[ch]*invN;
  float var = sums[32+ch]*invN - mu*mu;
  float val = (x[i]-mu)*rsqrtf(var + 1e-5f)*g[ch] + b[ch];
  x[i] = val;
  xbf[i] = f2bf(val);
}

// ---------------- sampling head ----------------
__global__ void tsg_gumbel(float* __restrict__ g, u32 k0, u32 k1, int n){
  int i = blockIdx.x*blockDim.x + threadIdx.x;
  int half = (n+1) >> 1;
  if (i >= half) return;
  u32 w0, w1;
  threefry2x32(k0, k1, (u32)i, (u32)(i+half), &w0, &w1);
  g[i] = bits_to_gumbel(w0);
  if (i + half < n) g[i+half] = bits_to_gumbel(w1);
}

__global__ void tsg_remlogits(const float* __restrict__ x64, const float* __restrict__ rw,
  const float* __restrict__ rb, const float* __restrict__ mask, const float* __restrict__ g1,
  float* __restrict__ rl_out, u32* __restrict__ rlmax, u64* __restrict__ a1pack){
  int t = blockIdx.x*blockDim.x + threadIdx.x;
  int node = t >> 5, lane = t & 31;
  if (node >= NNODES) return;
  float2 xv = ((const float2*)(x64 + (size_t)node*64))[lane];
  float2 wv = ((const float2*)rw)[lane];
  float p = xv.x*wv.x + xv.y*wv.y;
  #pragma unroll
  for (int off = 16; off; off >>= 1) p += __shfl_xor(p, off, 32);
  if (lane == 0){
    float m = mask[node];
    float rm;
    if (node < 15)
      rm = (m == 0.f) ? -__builtin_inff() : ((m == -__builtin_inff()) ? 0.f : m);
    else rm = m;
    float rl = p + rb[0] + rm;
    rl_out[node] = rl;
    atomicMax(rlmax, encf(rl));
    u64 pk = ((u64)encf(rl + g1[node]) << 32) | (u32)node;
    atomicMax(a1pack, pk);
  }
}

__global__ void tsg_projk(const float* __restrict__ x64, const float* __restrict__ aw,
  const float* __restrict__ ab, const u64* __restrict__ a1p, float* __restrict__ proj){
  int o = threadIdx.x;                   // 64 threads
  u32 a1 = (u32)(*a1p & 0xFFFFFFFFull);
  const float* xr = x64 + (size_t)a1*64;
  float s = ab[o];
  #pragma unroll 8
  for (int j = 0; j < 64; j++) s = fmaf(aw[o*64 + j], xr[j], s);
  proj[o] = tanhf(s);
}

__global__ void tsg_sumexp(const float* __restrict__ vals, const u32* __restrict__ vmax,
  float* __restrict__ vsum, int n){
  int i = blockIdx.x*blockDim.x + threadIdx.x;
  if (i >= n) return;
  atomicAdd(vsum, __expf(vals[i] - decf(*vmax)));
}

__global__ void tsg_newlogits(const float* __restrict__ x64, const float* __restrict__ proj,
  const float* __restrict__ mask, const float* __restrict__ g2, const u64* __restrict__ a1p,
  float* __restrict__ nl_out, u32* __restrict__ nlmax, u64* __restrict__ a2pack){
  int t = blockIdx.x*blockDim.x + threadIdx.x;
  int node = t >> 5, lane = t & 31;
  if (node >= NNODES) return;
  float2 xv = ((const float2*)(x64 + (size_t)node*64))[lane];
  float2 pv = ((const float2*)proj)[lane];
  float p = xv.x*pv.x + xv.y*pv.y;
  #pragma unroll
  for (int off = 16; off; off >>= 1) p += __shfl_xor(p, off, 32);
  if (lane == 0){
    u32 a1 = (u32)(*a1p & 0xFFFFFFFFull);
    float m2 = (node == (int)a1) ? 0.f : mask[node];
    float nl = p + m2;
    nl_out[node] = nl;
    atomicMax(nlmax, encf(nl));
    u64 pk = ((u64)encf(nl + g2[node]) << 32) | (u32)node;
    atomicMax(a2pack, pk);
  }
}

__global__ void tsg_final(float* out, const u64* a1p, const u64* a2p,
  const u32* rlmax, const float* rlsum, const u32* nlmax, const float* nlsum){
  u32 a1 = (u32)(*a1p & 0xFFFFFFFFull);
  u32 a2 = (u32)(*a2p & 0xFFFFFFFFull);
  out[2*NNODES+0] = (float)a1;
  out[2*NNODES+1] = (float)a2;
  out[2*NNODES+2] = out[a1]          - decf(*rlmax) - __logf(*rlsum);
  out[2*NNODES+3] = out[NNODES + a2] - decf(*nlmax) - __logf(*nlsum);
}

// ---------------- host orchestration ----------------
extern "C" void kernel_launch(void* const* d_in, const int* in_sizes, int n_in,
                              void* d_out, int out_size, void* d_ws, size_t ws_size,
                              hipStream_t stream){
  (void)in_sizes; (void)n_in; (void)out_size; (void)ws_size;
  const float* node_type   = (const float*)d_in[0];
  const float* requests    = (const float*)d_in[1];
  const float* latency     = (const float*)d_in[2];
  const float* active_mask = (const float*)d_in[3];
  const int*   ei          = (const int*)d_in[4];
  const int* src = ei;
  const int* dst = ei + NEDGES;
  const float* q1_w=(const float*)d_in[5],  *q1_b=(const float*)d_in[6];
  const float* k1_w=(const float*)d_in[7],  *k1_b=(const float*)d_in[8];
  const float* v1_w=(const float*)d_in[9],  *v1_b=(const float*)d_in[10];
  const float* e1_w=(const float*)d_in[11];
  const float* skip1_w=(const float*)d_in[12], *skip1_b=(const float*)d_in[13];
  const float* beta1_w=(const float*)d_in[14];
  const float* transf1_w=(const float*)d_in[15], *transf1_b=(const float*)d_in[16];
  const float* bn1_g=(const float*)d_in[17], *bn1_b=(const float*)d_in[18];
  const float* qL_w=(const float*)d_in[19], *qL_b=(const float*)d_in[20];
  const float* kL_w=(const float*)d_in[21], *kL_b=(const float*)d_in[22];
  const float* vL_w=(const float*)d_in[23], *vL_b=(const float*)d_in[24];
  const float* eL_w=(const float*)d_in[25];
  const float* skipL_w=(const float*)d_in[26], *skipL_b=(const float*)d_in[27];
  const float* betaL_w=(const float*)d_in[28];
  const float* transfL_w=(const float*)d_in[29], *transfL_b=(const float*)d_in[30];
  const float* bnL_g=(const float*)d_in[31], *bnL_b=(const float*)d_in[32];
  const float* lin1_w=(const float*)d_in[33], *lin1_b=(const float*)d_in[34];
  const float* lin2_w=(const float*)d_in[35], *lin2_b=(const float*)d_in[36];
  const float* rem_w=(const float*)d_in[37], *rem_b=(const float*)d_in[38];
  const float* addp_w=(const float*)d_in[39], *addp_b=(const float*)d_in[40];
  float* outf = (float*)d_out;

  // workspace carve-up (256B aligned)
  char* wsp = (char*)d_ws;
  auto take = [&](size_t bytes)->void*{
    void* p = (void*)wsp; wsp += (bytes + 255) & ~(size_t)255; return p;
  };
  float* q    = (float*)take((size_t)NNODES*128*4);
  float* kk   = (float*)take((size_t)NNODES*128*4);
  float* v    = (float*)take((size_t)NNODES*128*4);
  float* skip = (float*)take((size_t)NNODES*128*4);
  float* agg  = (float*)take((size_t)NNODES*128*4);
  float* x128 = (float*)take((size_t)NNODES*128*4);
  float* x32  = (float*)take((size_t)NNODES*32*4);
  float* x64  = (float*)take((size_t)NNODES*64*4);
  u16*   xbf  = (u16*)take((size_t)NNODES*128*2);
  u16*   xbf2 = (u16*)take((size_t)NNODES*128*2);
  float* alpha= (float*)take((size_t)NEDGES*4*4);
  float* exb  = (float*)take((size_t)NEDGES*4*4);
  u32*   amax = (u32*)take((size_t)NNODES*4*4);
  float* denom= (float*)take((size_t)NNODES*4*4);
  u16*   wbf  = (u16*)take(16384*2);
  float* g1   = (float*)take((size_t)NNODES*4);
  float* g2   = (float*)take((size_t)NNODES*4);
  float* bns  = (float*)take(64*4);
  u64*   a1pack = (u64*)take(512);
  u64*   a2pack = a1pack + 1;
  u32*   rlmax  = (u32*)(a1pack + 2);
  u32*   nlmax  = rlmax + 1;
  float* rlsum  = (float*)(nlmax + 1);
  float* nlsum  = rlsum + 1;
  float* projv  = nlsum + 1;   // 64 floats

  auto zf = [&](float* p, size_t n){
    tsg_zero_f<<<(unsigned)((n+255)/256),256,0,stream>>>(p, n); };
  auto zu = [&](u32* p, size_t n){
    tsg_zero_u<<<(unsigned)((n+255)/256),256,0,stream>>>(p, n); };
  auto cvt = [&](const float* s, u16* d, int n){
    tsg_f2bf_k<<<(unsigned)((n+255)/256),256,0,stream>>>(s, d, n); };
  auto gemm = [&](const u16* A, const u16* W, const float* bias, float* out,
                  u16* outbf, int K, int Nc, int relu){
    unsigned waves = (NNODES/16) * (unsigned)(Nc/16);
    tsg_gemm<<<(waves+7)/8,256,0,stream>>>(A, W, bias, out, outbf, NNODES, K, Nc, relu);
  };

  // edge attention + gate + transf + BN, shared by all 4 layers
  auto tconv_tail = [&](const float* ew, const float* bw, const float* tw,
                        const float* tb, const float* bg, const float* bb){
    zf(agg, (size_t)NNODES*128); zf(denom, (size_t)NNODES*4); zu(amax, (size_t)NNODES*4);
    tsg_alpha  <<<(NEDGES*4+255)/256,256,0,stream>>>(src,dst,latency,q,kk,ew,v,alpha,amax);
    tsg_expnorm<<<(NEDGES*4+255)/256,256,0,stream>>>(dst,alpha,amax,denom,exb);
    tsg_msg    <<<(NEDGES*32)/256,   256,0,stream>>>(src,dst,latency,v,ew,exb,denom,agg);
    tsg_gate   <<<(NNODES*32+255)/256,256,0,stream>>>(agg,skip,bw,xbf);
    cvt(tw, wbf, 32*128);
    gemm(xbf, wbf, tb, x32, nullptr, 128, 32, 1);       // relu(x @ transf.T + b)
    zf(bns, 64);
    tsg_bnstats<<<480,256,0,stream>>>(x32, bns);
    tsg_bnapply<<<(NNODES*32+255)/256,256,0,stream>>>(x32, bns, bg, bb, xbf);
  };

  // ---- layer 1 (FIN=2 projections on VALU) ----
  tsg_l1proj<<<(NNODES*128+255)/256,256,0,stream>>>(node_type,requests,
      q1_w,q1_b,k1_w,k1_b,v1_w,v1_b,skip1_w,skip1_b,q,kk,v,skip);
  tconv_tail(e1_w, beta1_w, transf1_w, transf1_b, bn1_g, bn1_b);

  // ---- layers 2..4 (WMMA projections, K=32 -> single wmma per tile) ----
  for (int li = 0; li < 3; li++){
    cvt(qL_w   +li*128*32, wbf, 128*32); gemm(xbf, wbf, qL_b   +li*128, q,    nullptr, 32, 128, 0);
    cvt(kL_w   +li*128*32, wbf, 128*32); gemm(xbf, wbf, kL_b   +li*128, kk,   nullptr, 32, 128, 0);
    cvt(vL_w   +li*128*32, wbf, 128*32); gemm(xbf, wbf, vL_b   +li*128, v,    nullptr, 32, 128, 0);
    cvt(skipL_w+li*128*32, wbf, 128*32); gemm(xbf, wbf, skipL_b+li*128, skip, nullptr, 32, 128, 0);
    tconv_tail(eL_w+li*128, betaL_w+li*384, transfL_w+li*32*128, transfL_b+li*32,
               bnL_g+li*32, bnL_b+li*32);
  }

  // ---- MLP ----
  cvt(lin1_w, wbf, 128*32); gemm(xbf,  wbf, lin1_b, x128, xbf2, 32, 128, 1);
  cvt(lin2_w, wbf, 64*128); gemm(xbf2, wbf, lin2_b, x64,  nullptr, 128, 64, 1);

  // ---- sampling head (threefry2x32, key 42) ----
  zu((u32*)a1pack, 8);   // a1pack, a2pack, rlmax, nlmax, rlsum, nlsum
  u32 s0,s1,t0,t1;
  threefry2x32(0u, 42u, 0u, 2u, &s0, &s1);
  threefry2x32(0u, 42u, 1u, 3u, &t0, &t1);
  int half = (NNODES+1)/2;
  tsg_gumbel<<<(half+255)/256,256,0,stream>>>(g1, s0, t0, NNODES);
  tsg_gumbel<<<(half+255)/256,256,0,stream>>>(g2, s1, t1, NNODES);
  tsg_remlogits<<<(NNODES*32+255)/256,256,0,stream>>>(x64, rem_w, rem_b, active_mask,
                                                      g1, outf, rlmax, a1pack);
  tsg_projk<<<1,64,0,stream>>>(x64, addp_w, addp_b, a1pack, projv);
  tsg_sumexp<<<(NNODES+255)/256,256,0,stream>>>(outf, rlmax, rlsum, NNODES);
  tsg_newlogits<<<(NNODES*32+255)/256,256,0,stream>>>(x64, projv, active_mask, g2,
                                                      a1pack, outf+NNODES, nlmax, a2pack);
  tsg_sumexp<<<(NNODES+255)/256,256,0,stream>>>(outf+NNODES, nlmax, nlsum, NNODES);
  tsg_final<<<1,1,0,stream>>>(outf, a1pack, a2pack, rlmax, rlsum, nlmax, nlsum);
}
